// SelfAttention_11063835754664
// MI455X (gfx1250) — matrix operand order repaired and checked by reading, not compiled, not run on hardware
//
#include <hip/hip_runtime.h>
#include <hip/hip_bf16.h>

typedef __bf16 bf16;
typedef __attribute__((ext_vector_type(8)))  __bf16 v8bf;
typedef __attribute__((ext_vector_type(16))) __bf16 v16bf;
typedef __attribute__((ext_vector_type(8)))  float  v8f;

#define D_MODEL  1024
#define N_HEADS  16
#define HEAD_DIM 64
#define SEQ_T    2048
#define BATCH    2

static __device__ __forceinline__ v8f wmma_bf16(v16bf a, v16bf b, v8f c) {
  // 8 args: (neg_a, A, neg_b, B, c_mod, C, reuse_a, reuse_b)
  return __builtin_amdgcn_wmma_f32_16x16x32_bf16(false, a, false, b, (short)0, c, false, false);
}

// Build a v16bf fragment from two 16-byte (8-half) aligned loads.
static __device__ __forceinline__ v16bf ld_frag2(const bf16* p0, const bf16* p1) {
  v16bf r;
  *(v8bf*)&r       = *(const v8bf*)p0;
  *((v8bf*)&r + 1) = *(const v8bf*)p1;
  return r;
}

// ---------------------------------------------------------------------------
// bf16-WMMA GEMM: C[M,N] = A[M,K] * B[K,N] (+bias)
// Block tile 128x128, 8 waves as 4(M)x2(N); wave tile 32x64 (2x4 WMMA tiles).
// Register-staged double buffering: global loads for tile k+1 are issued
// right after the barrier and overlap the 8 WMMAs consuming tile k from LDS.
// EPI==0: scatter bf16 q/k/v into [B,H,T,hd] buffers.  EPI==1: fp32 out.
// ---------------------------------------------------------------------------
template <typename TIN, int EPI>
__global__ __launch_bounds__(256)
void gemm_wmma_kernel(const TIN* __restrict__ A, const float* __restrict__ Bw,
                      const float* __restrict__ bias,
                      int M, int N, int K,
                      float* __restrict__ outf,
                      bf16* __restrict__ qb, bf16* __restrict__ kb,
                      bf16* __restrict__ vb)
{
  __shared__ bf16 As[128 * 40];   // [m][k], rows padded to 40 halves (80B -> 16B aligned)
  __shared__ bf16 Bs[128 * 40];   // transposed: [n][k]

  const int tid  = threadIdx.x;
  const int lane = tid & 31;
  const int wave = tid >> 5;
  const int g    = lane >> 4;     // half-wave select
  const int l16  = lane & 15;
  const int wm   = wave >> 1;     // 0..3  (M)
  const int wn   = wave & 1;      // 0..1  (N)
  const int m0   = blockIdx.y * 128;
  const int n0   = blockIdx.x * 128;

  const v8f vz = {0.f, 0.f, 0.f, 0.f, 0.f, 0.f, 0.f, 0.f};
  v8f acc[2][4] = {{vz, vz, vz, vz}, {vz, vz, vz, vz}};

  const int ar = tid >> 1;          // 0..127
  const int ac = (tid & 1) * 16;    // 0,16

  float aReg[16], bReg[16];
  auto load_tile = [&](int k0) {
    const TIN* ap = A + (size_t)(m0 + ar) * K + k0 + ac;
#pragma unroll
    for (int i = 0; i < 16; ++i) aReg[i] = (float)ap[i];
#pragma unroll
    for (int i = 0; i < 16; ++i) {
      int idx = tid + 256 * i;     // 0..4095
      int bk  = idx >> 7;          // 0..31
      int bn  = idx & 127;
      bReg[i] = Bw[(size_t)(k0 + bk) * N + n0 + bn];
    }
    if (k0 + 32 < K) {             // prefetch one more tile ahead (global_prefetch_b8)
      __builtin_prefetch(ap + 32, 0, 3);
      __builtin_prefetch(Bw + (size_t)(k0 + 32 + (tid >> 3)) * N + n0 + (tid & 7) * 16, 0, 3);
    }
  };

  load_tile(0);
  for (int k0 = 0; k0 < K; k0 += 32) {
    // Commit staged registers to LDS (bf16 conversion here).
#pragma unroll
    for (int i = 0; i < 16; ++i) As[ar * 40 + ac + i] = (bf16)aReg[i];
#pragma unroll
    for (int i = 0; i < 16; ++i) {
      int idx = tid + 256 * i;
      int bk  = idx >> 7;
      int bn  = idx & 127;
      Bs[bn * 40 + bk] = (bf16)bReg[i];
    }
    __syncthreads();

    if (k0 + 32 < K) load_tile(k0 + 32);   // overlaps with WMMA below (LOADcnt)

    v16bf afr[2], bfr[4];
#pragma unroll
    for (int mt = 0; mt < 2; ++mt) {
      const bf16* ap = &As[(wm * 32 + mt * 16 + l16) * 40];
      afr[mt] = ld_frag2(ap + g * 8, ap + 16 + g * 8);   // A layout: k in {g*8..}{16+g*8..}
    }
#pragma unroll
    for (int nt = 0; nt < 4; ++nt) {
      const bf16* bp = &Bs[(wn * 64 + nt * 16 + l16) * 40];
      bfr[nt] = ld_frag2(bp + g * 8, bp + 16 + g * 8);   // B layout: k in {g*8..}{16+g*8..}
    }
#pragma unroll
    for (int mt = 0; mt < 2; ++mt)
#pragma unroll
      for (int nt = 0; nt < 4; ++nt)
        acc[mt][nt] = wmma_bf16(afr[mt], bfr[nt], acc[mt][nt]);
    __syncthreads();
  }

  // Epilogue. C/D layout: lane(g,l16) holds rows M=g*8+r, col N=l16.
#pragma unroll
  for (int mt = 0; mt < 2; ++mt) {
#pragma unroll
    for (int nt = 0; nt < 4; ++nt) {
      const int nn = n0 + wn * 64 + nt * 16 + l16;
      const float bv = bias[nn];
#pragma unroll
      for (int r = 0; r < 8; ++r) {
        const int mm = m0 + wm * 32 + mt * 16 + g * 8 + r;
        const float v = acc[mt][nt][r] + bv;
        if (EPI == 0) {
          const int which = nn >> 10;          // / D_MODEL
          const int hh    = (nn & (D_MODEL - 1)) >> 6;
          const int d     = nn & (HEAD_DIM - 1);
          const int bb    = mm / SEQ_T;
          const int tt    = mm % SEQ_T;
          bf16* dst = (which == 0) ? qb : ((which == 1) ? kb : vb);
          dst[(((size_t)bb * N_HEADS + hh) * SEQ_T + tt) * HEAD_DIM + d] = (bf16)v;
        } else {
          outf[(size_t)mm * N + nn] = v;
        }
      }
    }
  }
}

// ---------------------------------------------------------------------------
// Flash attention, S^T = K*Q^T formulation (q across lanes -> cheap softmax:
// per-lane max/sum + one __shfl_xor(16) pair combine).
// Block = one (b,h) and 128 q rows; each of 8 waves owns 16 q rows.
// kv processed in 32-wide chunks: 4 WMMA for S^T, 4 WMMA for y^T.
// V staging is register-double-buffered; mask ops only on diagonal chunks.
// ---------------------------------------------------------------------------
__global__ __launch_bounds__(256)
void flash_attn_kernel(const bf16* __restrict__ Qb, const bf16* __restrict__ Kb,
                       const bf16* __restrict__ Vb, bf16* __restrict__ Yb)
{
  __shared__ bf16 VT[HEAD_DIM * 40];     // V^T staged: [hd][kv(32)] padded rows
  __shared__ bf16 Pl[8 * 16 * 40];       // per-wave P: [q(16)][kv(32)] padded rows

  const int tid  = threadIdx.x;
  const int lane = tid & 31;
  const int wave = tid >> 5;
  const int g    = lane >> 4;
  const int l16  = lane & 15;

  const int nqblk = SEQ_T / 128;         // 16
  const int qblk  = blockIdx.x % nqblk;
  const int bh    = blockIdx.x / nqblk;  // b*H + h
  const int h     = bh % N_HEADS;
  const int b     = bh / N_HEADS;

  const bf16* Qp = Qb + (size_t)bh * SEQ_T * HEAD_DIM;
  const bf16* Kp = Kb + (size_t)bh * SEQ_T * HEAD_DIM;
  const bf16* Vp = Vb + (size_t)bh * SEQ_T * HEAD_DIM;

  const int q0 = qblk * 128 + wave * 16;

  // Preload Q^T as two B-operand fragments (hd chunks 0-31, 32-63); reused all loop.
  v16bf bq[2];
#pragma unroll
  for (int c = 0; c < 2; ++c) {
    const bf16* qp = Qp + (size_t)(q0 + l16) * HEAD_DIM + c * 32;
    bq[c] = ld_frag2(qp + g * 8, qp + 16 + g * 8);
  }

  const v8f vz = {0.f, 0.f, 0.f, 0.f, 0.f, 0.f, 0.f, 0.f};
  v8f o[4] = {vz, vz, vz, vz};           // y^T accumulators, hd tiles 0..3
  const float NEG_INF = -__builtin_inff();
  float mrun = NEG_INF;
  float lrun = 0.f;
  bf16* pw = &Pl[wave * 16 * 40];

  const int kv_end = qblk * 128 + 128;   // block-uniform causal bound

  bf16 vReg[8];
  auto load_v = [&](int kv0) {
#pragma unroll
    for (int i = 0; i < 8; ++i) {
      int idx = tid + 256 * i;           // 0..2047
      int kv  = idx >> 6;
      int d   = idx & 63;
      vReg[i] = Vp[(size_t)(kv0 + kv) * HEAD_DIM + d];
    }
  };

  load_v(0);
  for (int kv0 = 0; kv0 < kv_end; kv0 += 32) {
    __syncthreads();                     // previous chunk's VT readers done
#pragma unroll
    for (int i = 0; i < 8; ++i) {        // commit staged V, transposed
      int idx = tid + 256 * i;
      int kv  = idx >> 6;
      int d   = idx & 63;
      VT[d * 40 + kv] = vReg[i];
    }
    __syncthreads();

    if (kv0 + 32 < kv_end) {
      load_v(kv0 + 32);                  // overlaps with WMMA/softmax below
      __builtin_prefetch(Kp + (size_t)(kv0 + 32 + lane) * HEAD_DIM, 0, 3);
    }

    if (kv0 <= q0 + 15) {                // wave-uniform causal skip
      // S^T chunk: two 16(kv) x 16(q) tiles, contraction over hd=64.
      v8f st[2];
#pragma unroll
      for (int kvt = 0; kvt < 2; ++kvt) {
        const bf16* kp = Kp + (size_t)(kv0 + kvt * 16 + l16) * HEAD_DIM;
        v16bf ak0 = ld_frag2(kp + g * 8,      kp + 16 + g * 8);
        v16bf ak1 = ld_frag2(kp + 32 + g * 8, kp + 48 + g * 8);
        v8f s = vz;
        s = wmma_bf16(ak0, bq[0], s);
        s = wmma_bf16(ak1, bq[1], s);
        st[kvt] = s;
      }

      // Scale (+ mask only on diagonal chunks) and tile max.
      float mt = NEG_INF;
      if (kv0 + 31 <= q0) {              // fully unmasked interior chunk (wave-uniform)
#pragma unroll
        for (int kvt = 0; kvt < 2; ++kvt)
#pragma unroll
          for (int r = 0; r < 8; ++r) {
            const float sv = st[kvt][r] * 0.125f;   // 1/sqrt(64)
            st[kvt][r] = sv;
            mt = fmaxf(mt, sv);
          }
      } else {                            // diagonal chunk: apply causal mask
#pragma unroll
        for (int kvt = 0; kvt < 2; ++kvt)
#pragma unroll
          for (int r = 0; r < 8; ++r) {
            const int kvi = kv0 + kvt * 16 + g * 8 + r;
            float sv = st[kvt][r] * 0.125f;
            sv = (kvi <= q0 + l16) ? sv : NEG_INF;
            st[kvt][r] = sv;
            mt = fmaxf(mt, sv);
          }
      }
      mt = fmaxf(mt, __shfl_xor(mt, 16));
      const float mnew = fmaxf(mrun, mt);
      const float scl  = __expf(mrun - mnew);
      float ls = 0.f;
#pragma unroll
      for (int kvt = 0; kvt < 2; ++kvt)
#pragma unroll
        for (int r = 0; r < 8; ++r) {
          const float p = __expf(st[kvt][r] - mnew);
          st[kvt][r] = p;
          ls += p;
        }
      ls += __shfl_xor(ls, 16);
      lrun = lrun * scl + ls;
      mrun = mnew;
#pragma unroll
      for (int t = 0; t < 4; ++t)
#pragma unroll
        for (int r = 0; r < 8; ++r) o[t][r] *= scl;

      // P^T -> wave-private LDS as [q][kv] (contiguous store and reload).
#pragma unroll
      for (int kvt = 0; kvt < 2; ++kvt) {
        v8bf pv;
#pragma unroll
        for (int r = 0; r < 8; ++r) pv[r] = (bf16)st[kvt][r];
        *(v8bf*)(pw + l16 * 40 + kvt * 16 + g * 8) = pv;
      }

      // y^T += V^T(chunk) * P^T : 4 hd tiles, contraction kv=32.
      const bf16* pp = pw + l16 * 40;
      v16bf bp_ = ld_frag2(pp + g * 8, pp + 16 + g * 8);
#pragma unroll
      for (int t = 0; t < 4; ++t) {
        const bf16* vp = &VT[(t * 16 + l16) * 40];
        v16bf av = ld_frag2(vp + g * 8, vp + 16 + g * 8);
        o[t] = wmma_bf16(av, bp_, o[t]);
      }
    }
  }

  // Normalize and write Y (bf16, [B*T, C] layout), 16B contiguous stores.
  const float inv = 1.0f / lrun;
#pragma unroll
  for (int t = 0; t < 4; ++t) {
    v8bf yv;
#pragma unroll
    for (int r = 0; r < 8; ++r) yv[r] = (bf16)(o[t][r] * inv);
    bf16* yp = Yb + ((size_t)b * SEQ_T + q0 + l16) * D_MODEL + h * HEAD_DIM + t * 16 + g * 8;
    *(v8bf*)yp = yv;
  }
}

// ---------------------------------------------------------------------------
extern "C" void kernel_launch(void* const* d_in, const int* in_sizes, int n_in,
                              void* d_out, int out_size, void* d_ws, size_t ws_size,
                              hipStream_t stream) {
  (void)in_sizes; (void)n_in; (void)out_size; (void)ws_size;
  const float* x      = (const float*)d_in[0];
  const float* W_attn = (const float*)d_in[1];
  const float* b_attn = (const float*)d_in[2];
  const float* W_proj = (const float*)d_in[3];
  const float* b_proj = (const float*)d_in[4];
  float* out = (float*)d_out;

  const size_t bhte = (size_t)BATCH * N_HEADS * SEQ_T * HEAD_DIM;  // 4,194,304
  bf16* qb = (bf16*)d_ws;
  bf16* kb = qb + bhte;
  bf16* vb = kb + bhte;
  bf16* yb = vb + bhte;   // [B*T, C] bf16; total ws use = 32 MB

  const int M = BATCH * SEQ_T;       // 4096
  dim3 blk(256);

  // 1) qkv = x @ W_attn + b_attn, scattered into bf16 Q/K/V [B,H,T,hd]
  gemm_wmma_kernel<float, 0><<<dim3(3 * D_MODEL / 128, M / 128), blk, 0, stream>>>(
      x, W_attn, b_attn, M, 3 * D_MODEL, D_MODEL, nullptr, qb, kb, vb);

  // 2) causal flash attention -> Y (bf16)
  flash_attn_kernel<<<dim3(BATCH * N_HEADS * (SEQ_T / 128)), blk, 0, stream>>>(
      qb, kb, vb, yb);

  // 3) out = Y @ W_proj + b_proj (fp32)
  gemm_wmma_kernel<bf16, 1><<<dim3(D_MODEL / 128, M / 128), blk, 0, stream>>>(
      yb, W_proj, b_proj, M, D_MODEL, D_MODEL, out, nullptr, nullptr, nullptr);
}
